// AggregateEmbedding_80556406604255
// MI455X (gfx1250) — compile-verified, matrix operand order checked by reading, not run on hardware
//
#include <hip/hip_runtime.h>
#include <hip/hip_bf16.h>
#include <hip/hip_fp16.h>

// ---------------------------------------------------------------------------
// MI455X (gfx1250, wave32) implementation.
//   E=128, S=100, B=512, N=EG=51200, TIME_NUM=50, MAX_TIME=1.0
// All GEMMs via v_wmma_f32_16x16x32_f16 (f16 in, f32 accumulate).
// ---------------------------------------------------------------------------

typedef __attribute__((ext_vector_type(16))) _Float16 v16h;
typedef __attribute__((ext_vector_type(8)))  _Float16 v8h;
typedef __attribute__((ext_vector_type(8)))  float    v8f;

#define E_DIM 128
#define S_LEN 100
#define B_CAS 512
#define N_NODE 51200
#define EG_EDGE 51200

__device__ __forceinline__ int time_bucket(float t) {
  int i = (int)(t * 50.0f);
  return i < 0 ? 0 : (i > 49 ? 49 : i);
}

// Fast transcendentals: gfx1250 has V_TANH_F32 (single trans op).
__device__ __forceinline__ float fast_tanh(float x) {
#if __has_builtin(__builtin_amdgcn_tanhf)
  return __builtin_amdgcn_tanhf(x);
#else
  float e = __expf(2.0f * x);
  return 1.0f - 2.0f * __builtin_amdgcn_rcpf(e + 1.0f);
#endif
}
__device__ __forceinline__ float fast_sigmoid(float x) {
#if __has_builtin(__builtin_amdgcn_tanhf)
  return 0.5f + 0.5f * __builtin_amdgcn_tanhf(0.5f * x);
#else
  return __builtin_amdgcn_rcpf(1.0f + __expf(-x));
#endif
}

__device__ __forceinline__ v8f wmma16(v16h a, v16h b, v8f c) {
  // D = A(16x32 f16) x B(32x16 f16) + C(16x16 f32)
  return __builtin_amdgcn_wmma_f32_16x16x32_f16(false, a, false, b, (short)0, c,
                                                false, false);
}

// A fragment (16x32, 16-bit): lane (m = lane&15, half = lane>>4) holds
//   k = k0 + 8*half + [0..7]   and   k = k0 + 16 + 8*half + [0..7]
// Same per-lane pattern serves the B fragment when W is row-major W[n][k]
// (lane n = lane&15, contiguous k runs).
__device__ __forceinline__ v16h frag_f16(const _Float16* rowbase, int k0, int half) {
  const _Float16* p = rowbase + k0 + 8 * half;
  v8h c0 = *(const v8h*)(p);
  v8h c1 = *(const v8h*)(p + 16);
  v16h r;
#pragma unroll
  for (int i = 0; i < 8; ++i) { r[i] = c0[i]; r[i + 8] = c1[i]; }
  return r;
}

__device__ __forceinline__ v16h frag_f32(const float* rowbase, int k0, int half,
                                         float scale) {
  const float* p = rowbase + k0 + 8 * half;
  v16h r;
#pragma unroll
  for (int i = 0; i < 8; ++i) {
    r[i]     = (_Float16)(p[i] * scale);
    r[i + 8] = (_Float16)(p[16 + i] * scale);
  }
  return r;
}

// ---------------------------------------------------------------------------
// Kernel: f32 -> f16 weight pack (row-major layout preserved)
// ---------------------------------------------------------------------------
__global__ void f32_to_f16_kernel(const float* __restrict__ src,
                                  _Float16* __restrict__ dst, int n) {
  int i = blockIdx.x * blockDim.x + threadIdx.x;
  if (i < n) dst[i] = (_Float16)src[i];
}

// ---------------------------------------------------------------------------
// Kernel: x = static_emb[cas_history] + time_emb[bucket(times)] + pos_emb[s]
// written as f16 (B*S, E)
// ---------------------------------------------------------------------------
__global__ void build_x_kernel(const int* __restrict__ cas_history,
                               const float* __restrict__ cas_times,
                               const float* __restrict__ static_emb,
                               const float* __restrict__ time_emb,
                               const float* __restrict__ pos_emb,
                               _Float16* __restrict__ x16) {
  int tid = blockIdx.x * blockDim.x + threadIdx.x;
  if (tid >= B_CAS * S_LEN * E_DIM) return;
  int e  = tid & (E_DIM - 1);
  int bs = tid >> 7;
  int s  = bs % S_LEN;
  int uid = cas_history[bs];
  int tb  = time_bucket(cas_times[bs]);
  float v = static_emb[(size_t)uid * E_DIM + e] + time_emb[tb * E_DIM + e] +
            pos_emb[s * E_DIM + e];
  x16[tid] = (_Float16)v;
}

// ---------------------------------------------------------------------------
// Kernel: LSTM with length-masked state freeze.
// 32 blocks x 256 threads (8 waves). Block owns 16 batch rows.
// Wave w owns gate columns [64w, 64w+64) -> a single gate type (i/f/g/o).
// W_ih, W_hh staged f16 in LDS (256 KB); h recirculates via LDS f16.
// ---------------------------------------------------------------------------
__global__ void lstm_kernel(const _Float16* __restrict__ x16,
                            const _Float16* __restrict__ wih16,
                            const _Float16* __restrict__ whh16,
                            const float* __restrict__ b_ih,
                            const float* __restrict__ b_hh,
                            const int* __restrict__ lengths,
                            float* __restrict__ h_out) {
  extern __shared__ char smem[];
  _Float16* Wih  = (_Float16*)smem;               // 512*128 f16 = 128 KB
  _Float16* Whh  = Wih + 512 * 128;               // 128 KB
  float*    gates = (float*)(Whh + 512 * 128);    // 16*512 f32 = 32 KB
  _Float16* h16  = (_Float16*)(gates + 16 * 512); // 16*128 f16 = 4 KB
  float*    cf   = (float*)(h16 + 16 * 128);      // 16*128 f32 = 8 KB
  int*      lens = (int*)(cf + 16 * 128);         // 16 ints

  const int tid  = threadIdx.x;
  const int w    = tid >> 5;
  const int lane = tid & 31;
  const int half = lane >> 4;
  const int mr   = lane & 15;
  const int rb   = blockIdx.x * 16;

  {  // stage weights into LDS with b128 copies
    const uint4* s0 = (const uint4*)wih16;
    const uint4* s1 = (const uint4*)whh16;
    uint4* d0 = (uint4*)Wih;
    uint4* d1 = (uint4*)Whh;
    for (int i = tid; i < 512 * 128 * 2 / 16; i += 256) {
      d0[i] = s0[i];
      d1[i] = s1[i];
    }
  }
  for (int i = tid; i < 16 * 128; i += 256) {
    h16[i] = (_Float16)0.0f;
    cf[i]  = 0.0f;
  }
  if (tid < 16) {
    int L = lengths[rb + tid];
    lens[tid] = L < 1 ? 1 : L;
  }
  __syncthreads();

  float bsum[4];
#pragma unroll
  for (int nt = 0; nt < 4; ++nt) {
    int col = 64 * w + 16 * nt + mr;
    bsum[nt] = b_ih[col] + b_hh[col];
  }
  // wave-uniform gate type, forced scalar so the nonlinearity branch is SALU
  const bool is_g = (__builtin_amdgcn_readfirstlane(tid >> 6) == 2);
  const _Float16* xrow = x16 + (size_t)(rb + mr) * S_LEN * E_DIM;

  for (int t = 0; t < S_LEN; ++t) {
    // A fragments for x_t (global, L2-hot) and h_{t-1} (LDS)
    v16h ax[4], ah[4];
    const _Float16* xr = xrow + (size_t)t * E_DIM;
#pragma unroll
    for (int kt = 0; kt < 4; ++kt) ax[kt] = frag_f16(xr, 32 * kt, half);
#pragma unroll
    for (int kt = 0; kt < 4; ++kt) ah[kt] = frag_f16(h16 + mr * E_DIM, 32 * kt, half);

#pragma unroll
    for (int nt = 0; nt < 4; ++nt) {
      v8f acc = {};  // inline-0 C operand; bias added post-GEMM (no splat spill)
      const _Float16* wr_ih = Wih + (size_t)(64 * w + 16 * nt + mr) * E_DIM;
      const _Float16* wr_hh = Whh + (size_t)(64 * w + 16 * nt + mr) * E_DIM;
#pragma unroll
      for (int kt = 0; kt < 4; ++kt) {
        acc = wmma16(ax[kt], frag_f16(wr_ih, 32 * kt, half), acc);
        acc = wmma16(ah[kt], frag_f16(wr_hh, 32 * kt, half), acc);
      }
      // gate nonlinearity: this wave owns exactly one gate type
      if (is_g) {
#pragma unroll
        for (int r = 0; r < 8; ++r)
          gates[(r + 8 * half) * 512 + 64 * w + 16 * nt + mr] =
              fast_tanh(acc[r] + bsum[nt]);
      } else {
#pragma unroll
        for (int r = 0; r < 8; ++r)
          gates[(r + 8 * half) * 512 + 64 * w + 16 * nt + mr] =
              fast_sigmoid(acc[r] + bsum[nt]);
      }
    }
    __syncthreads();

    // elementwise cell update (length-masked freeze)
#pragma unroll
    for (int u = 0; u < 8; ++u) {
      int idx = tid + 256 * u;          // idx = row*128 + j
      int row = idx >> 7, j = idx & 127;
      if (t < lens[row]) {
        float ig = gates[row * 512 + j];
        float fg = gates[row * 512 + 128 + j];
        float gg = gates[row * 512 + 256 + j];
        float og = gates[row * 512 + 384 + j];
        float c  = fg * cf[idx] + ig * gg;
        cf[idx]  = c;
        h16[idx] = (_Float16)(og * fast_tanh(c));
      }
    }
    __syncthreads();
  }

  for (int i = tid; i < 16 * 128; i += 256)
    h_out[(size_t)(rb + (i >> 7)) * E_DIM + (i & 127)] = (float)h16[i];
}

// ---------------------------------------------------------------------------
// Kernel: tree message GEMM + graph-bucket accumulate.
// graph_id is contiguous (node//100), so segment_sum over nodes then graphs
// collapses to: msg[e] accumulated into bucket e_dst[e]/100.
// Each wave handles 16 edges: A = [static_emb[node_id[e_src]], time_emb[tb]]
// (16x256 f16, gathered+converted on the fly), B = W_msg (128x256 row-major).
// ---------------------------------------------------------------------------
__global__ void tree_msg_kernel(const float* __restrict__ static_emb,
                                const float* __restrict__ time_emb,
                                const int* __restrict__ node_id,
                                const int* __restrict__ e_src,
                                const int* __restrict__ e_dst,
                                const float* __restrict__ e_time,
                                const _Float16* __restrict__ wmsg16,
                                const float* __restrict__ b_msg,
                                float* __restrict__ gsum) {
  const int w    = threadIdx.x >> 5;
  const int lane = threadIdx.x & 31;
  const int half = lane >> 4;
  const int mr   = lane & 15;
  const int e0   = (blockIdx.x * 8 + w) * 16;

  const int eA   = e0 + mr;
  const int node = node_id[e_src[eA]];
  const int tb   = time_bucket(e_time[eA]);
  const float* srow = static_emb + (size_t)node * E_DIM;
  const float* trow = time_emb + (size_t)tb * E_DIM;

  // A fragments over K=256 (first 128: node emb, last 128: edge time emb)
  v16h a[8];
#pragma unroll
  for (int kt = 0; kt < 8; ++kt) {
    int ka = 32 * kt + 8 * half;
    int kb = ka + 16;
    const float* pa = (ka < 128) ? (srow + ka) : (trow + ka - 128);
    const float* pb = (kb < 128) ? (srow + kb) : (trow + kb - 128);
    v16h r;
#pragma unroll
    for (int i = 0; i < 8; ++i) {
      r[i]     = (_Float16)pa[i];
      r[i + 8] = (_Float16)pb[i];
    }
    a[kt] = r;
  }

  // graph bucket for the 8 C-rows this lane owns
  int gidx[8];
#pragma unroll
  for (int r = 0; r < 8; ++r) gidx[r] = e_dst[e0 + r + 8 * half] / 100;

#pragma unroll
  for (int nt = 0; nt < 8; ++nt) {
    v8f acc = {};
    const _Float16* wr = wmsg16 + (size_t)(16 * nt + mr) * 256;
#pragma unroll
    for (int kt = 0; kt < 8; ++kt)
      acc = wmma16(a[kt], frag_f16(wr, 32 * kt, half), acc);
    const int n = 16 * nt + mr;
    float bv = b_msg[n];
#pragma unroll
    for (int r = 0; r < 8; ++r) {
      float v = fmaxf(acc[r] + bv, 0.0f);  // relu(msg)
      unsafeAtomicAdd(&gsum[(size_t)gidx[r] * E_DIM + n], v);
    }
  }
}

// ---------------------------------------------------------------------------
// Kernel: out = relu([h | root/100 | leaf/100] @ W_trans.T + b_trans)
// 32 blocks x 8 waves; wave owns 16 output columns; K = 384 (12 k-tiles).
// ---------------------------------------------------------------------------
__global__ void final_kernel(const float* __restrict__ h,
                             const float* __restrict__ rsum,
                             const float* __restrict__ lsum,
                             const _Float16* __restrict__ wt16,
                             const float* __restrict__ b_trans,
                             float* __restrict__ out) {
  const int w    = threadIdx.x >> 5;
  const int lane = threadIdx.x & 31;
  const int half = lane >> 4;
  const int mr   = lane & 15;
  const int rb   = blockIdx.x * 16;
  const int n    = 16 * w + mr;

  v8f acc = {};
  const _Float16* wr = wt16 + (size_t)n * 384;
  const float* srcs[3] = { h + (size_t)(rb + mr) * E_DIM,
                           rsum + (size_t)(rb + mr) * E_DIM,
                           lsum + (size_t)(rb + mr) * E_DIM };
  const float scl[3] = { 1.0f, 0.01f, 0.01f };  // mean over 100 nodes/graph

#pragma unroll
  for (int kt = 0; kt < 12; ++kt) {
    int seg = kt >> 2;
    v16h a = frag_f32(srcs[seg], 32 * (kt & 3), half, scl[seg]);
    v16h b = frag_f16(wr, 32 * kt, half);
    acc = wmma16(a, b, acc);
  }
  float bv = b_trans[n];
#pragma unroll
  for (int r = 0; r < 8; ++r)
    out[(size_t)(rb + r + 8 * half) * E_DIM + n] = fmaxf(acc[r] + bv, 0.0f);
}

// ---------------------------------------------------------------------------
extern "C" void kernel_launch(void* const* d_in, const int* in_sizes, int n_in,
                              void* d_out, int out_size, void* d_ws, size_t ws_size,
                              hipStream_t stream) {
  const int*   cas_history = (const int*)  d_in[0];
  const float* cas_times   = (const float*)d_in[1];
  const int*   lengths     = (const int*)  d_in[2];
  const int*   node_root   = (const int*)  d_in[3];
  const int*   node_leaf   = (const int*)  d_in[4];
  const int*   esrc_root   = (const int*)  d_in[5];
  const int*   edst_root   = (const int*)  d_in[6];
  const float* etime_root  = (const float*)d_in[7];
  const int*   esrc_leaf   = (const int*)  d_in[8];
  const int*   edst_leaf   = (const int*)  d_in[9];
  const float* etime_leaf  = (const float*)d_in[10];
  const float* static_emb  = (const float*)d_in[11];
  const float* time_emb    = (const float*)d_in[12];
  const float* pos_emb     = (const float*)d_in[13];
  const float* W_ih        = (const float*)d_in[14];
  const float* W_hh        = (const float*)d_in[15];
  const float* b_ih        = (const float*)d_in[16];
  const float* b_hh        = (const float*)d_in[17];
  const float* W_msg_root  = (const float*)d_in[18];
  const float* b_msg_root  = (const float*)d_in[19];
  const float* W_msg_leaf  = (const float*)d_in[20];
  const float* b_msg_leaf  = (const float*)d_in[21];
  const float* W_trans     = (const float*)d_in[22];
  const float* b_trans     = (const float*)d_in[23];
  float* out = (float*)d_out;

  // workspace carve (~14.5 MB)
  char* ws = (char*)d_ws;
  size_t off = 0;
  auto carve = [&](size_t bytes) -> void* {
    void* p = ws + off;
    off = (off + bytes + 255) & ~(size_t)255;
    return p;
  };
  _Float16* x16   = (_Float16*)carve((size_t)B_CAS * S_LEN * E_DIM * 2);
  _Float16* wih16 = (_Float16*)carve(512 * 128 * 2);
  _Float16* whh16 = (_Float16*)carve(512 * 128 * 2);
  _Float16* wmr16 = (_Float16*)carve(128 * 256 * 2);
  _Float16* wml16 = (_Float16*)carve(128 * 256 * 2);
  _Float16* wt16  = (_Float16*)carve(128 * 384 * 2);
  float* h_out = (float*)carve((size_t)B_CAS * E_DIM * 4);
  float* rsum  = (float*)carve((size_t)B_CAS * E_DIM * 4);
  float* lsum  = (float*)carve((size_t)B_CAS * E_DIM * 4);

  hipMemsetAsync(rsum, 0, (size_t)B_CAS * E_DIM * 4, stream);
  hipMemsetAsync(lsum, 0, (size_t)B_CAS * E_DIM * 4, stream);

  // weight packs (f32 -> f16, layout preserved: B-fragments load contiguous k)
  f32_to_f16_kernel<<<(512 * 128 + 255) / 256, 256, 0, stream>>>(W_ih, wih16, 512 * 128);
  f32_to_f16_kernel<<<(512 * 128 + 255) / 256, 256, 0, stream>>>(W_hh, whh16, 512 * 128);
  f32_to_f16_kernel<<<(128 * 256 + 255) / 256, 256, 0, stream>>>(W_msg_root, wmr16, 128 * 256);
  f32_to_f16_kernel<<<(128 * 256 + 255) / 256, 256, 0, stream>>>(W_msg_leaf, wml16, 128 * 256);
  f32_to_f16_kernel<<<(128 * 384 + 255) / 256, 256, 0, stream>>>(W_trans, wt16, 128 * 384);

  // fused gather + time/pos embedding
  build_x_kernel<<<(B_CAS * S_LEN * E_DIM + 255) / 256, 256, 0, stream>>>(
      cas_history, cas_times, static_emb, time_emb, pos_emb, x16);

  // recurrent LSTM: 32 blocks of 16 batch rows, 307 KB LDS (<= 320 KB/WGP)
  size_t lds_bytes = (size_t)512 * 128 * 2 * 2   // Wih + Whh f16
                   + 16 * 512 * 4                // gates f32
                   + 16 * 128 * 2                // h f16
                   + 16 * 128 * 4                // c f32
                   + 16 * 4;                     // lengths
  lstm_kernel<<<32, 256, lds_bytes, stream>>>(x16, wih16, whh16, b_ih, b_hh,
                                              lengths, h_out);

  // tree GNN message GEMMs + graph accumulate (3200 edge-tiles / 8 waves)
  tree_msg_kernel<<<EG_EDGE / 16 / 8, 256, 0, stream>>>(
      static_emb, time_emb, node_root, esrc_root, edst_root, etime_root,
      wmr16, b_msg_root, rsum);
  tree_msg_kernel<<<EG_EDGE / 16 / 8, 256, 0, stream>>>(
      static_emb, time_emb, node_leaf, esrc_leaf, edst_leaf, etime_leaf,
      wml16, b_msg_leaf, lsum);

  // final fuse: relu([h | root | leaf] @ W_trans.T + b)
  final_kernel<<<32, 256, 0, stream>>>(h_out, rsum, lsum, wt16, b_trans, out);
}
